// Graph_weight_space_42666205118554
// MI455X (gfx1250) — matrix operand order repaired and checked by reading, not compiled, not run on hardware
//
#include <hip/hip_runtime.h>
#include <hip/hip_bf16.h>

typedef __attribute__((ext_vector_type(16))) __bf16 v16bf;
typedef __attribute__((ext_vector_type(8)))  __bf16 v8bf;
typedef __attribute__((ext_vector_type(2)))  __bf16 v2bf;
typedef __attribute__((ext_vector_type(8)))  float  v8f;
typedef __attribute__((ext_vector_type(4)))  float  v4f;

#define C_CH 128
#define HH 256
#define WWID 256
#define NBATCH 8
#define IMG (HH*WWID)            /* 65536 */
#define NPIX (NBATCH*IMG)        /* 524288 */

#define TP   64                  /* pixels per conv3 block  */
#define COLS 66                  /* TP + 2 halo columns     */
#define CHP  136                 /* padded channel stride (272B, 16B-aligned) */
#define NBLK3 8192               /* 8 * 256 * 4             */

/* workspace layout (bytes) */
#define OFF_PROW 0u                          /* 8*128*256 f32 = 1 MB  */
#define OFF_PCOL 1048576u                    /* 1 MB                  */
#define OFF_GATE 2097152u                    /* 8*256 f32 = 8 KB      */
#define OFF_WA1  2105344u                    /* 147456 bf16 = 288 KB  */
#define OFF_WA2  2400256u                    /* 16384 bf16 = 32 KB    */
#define OFF_SS   2433024u                    /* 256 f32 (scale,shift) */
#define OFF_BNP  2434048u                    /* 256*8192 f32 = 8 MB   */
#define OFF_HT   10822656u                   /* 524288*128 bf16 = 128 MB */

/* contiguous K order this code uses for its 16-bit A/B fragments (16x32):
   lane L holds M = L&15; element i (of v16h) -> K below. */
__device__ __forceinline__ int kfrag(int L, int i) {
  return ((L >> 4) << 4) | i;
}

/* ---- kernel 1: per-(b,c) row sums (over w) and col sums (over h) ---- */
__global__ void __launch_bounds__(256)
k_partials(const float* __restrict__ x, float* __restrict__ prow, float* __restrict__ pcol) {
  int bc = blockIdx.x;                       // b*128 + c
  int wave = threadIdx.x >> 5, L = threadIdx.x & 31;
  __shared__ __align__(16) float colLDS[8 * 256];
  const float* base0 = x + (size_t)bc * IMG;
  v4f colacc[2];
#pragma unroll
  for (int u = 0; u < 2; ++u) colacc[u] = (v4f){0.f, 0.f, 0.f, 0.f};
  for (int h = wave; h < HH; h += 8) {
    const v4f* row = (const v4f*)(base0 + h * WWID);
    float rsum = 0.f;
#pragma unroll
    for (int u = 0; u < 2; ++u) {
      v4f v = row[u * 32 + L];               // w = u*128 + L*4 .. +3
      colacc[u] += v;
      rsum += v.x + v.y + v.z + v.w;
    }
#pragma unroll
    for (int m = 16; m >= 1; m >>= 1) rsum += __shfl_xor(rsum, m, 32);
    if (L == 0) prow[(size_t)bc * HH + h] = rsum;
  }
#pragma unroll
  for (int u = 0; u < 2; ++u)
    *(v4f*)&colLDS[wave * 256 + u * 128 + L * 4] = colacc[u];
  __syncthreads();
  int w = threadIdx.x;
  float cs = 0.f;
  for (int wv = 0; wv < 8; ++wv) cs += colLDS[wv * 256 + w];
  pcol[(size_t)bc * WWID + w] = cs;
}

/* ---- kernel 2: xm, sign KNN (top-5 smallest dist, stable), gate ---- */
__global__ void __launch_bounds__(256)
k_gate(const float* __restrict__ prow, const float* __restrict__ pcol, float* __restrict__ gate) {
  int b = blockIdx.x, n = threadIdx.x;
  __shared__ float xmS[256], sS[256];
  float rs = 0.f, cs = 0.f;
  for (int c = 0; c < C_CH; ++c) {
    rs += prow[((size_t)(b * C_CH + c)) * 256 + n];
    cs += pcol[((size_t)(b * C_CH + c)) * 256 + n];
  }
  float xm = rs * (1.f / 32768.f) + cs * (1.f / 32768.f);  // /(C*W) and /(C*H)
  float v = xm * 0.5f;
  float s = v / fmaxf(fabsf(v), 1e-12f);
  xmS[n] = xm; sS[n] = s;
  __syncthreads();
  unsigned used[8] = {0, 0, 0, 0, 0, 0, 0, 0};
  float sum = 0.f;
  for (int kk = 0; kk < 5; ++kk) {
    float best = 3.4e38f; int bj = 0;
    for (int j = 0; j < 256; ++j) {
      if (used[j >> 5] & (1u << (j & 31))) continue;
      float d = s - sS[j]; d *= d;
      if (d < best) { best = d; bj = j; }   // strict < : lowest index on ties
    }
    used[bj >> 5] |= (1u << (bj & 31));
    sum += xmS[bj];
  }
  gate[b * 256 + n] = 1.f / (1.f + expf(-sum));
}

/* ---- kernel 3: pre-swizzle w1 / w2 into A-fragment-ordered bf16 ---- */
__global__ void k_prep(const float* __restrict__ w1, const float* __restrict__ w2,
                       __bf16* __restrict__ wA1, __bf16* __restrict__ wA2) {
  const int total1 = 8 * 36 * 512;   // 147456
  const int total2 = 8 * 4 * 512;    // 16384
  for (int e = blockIdx.x * blockDim.x + threadIdx.x; e < total1 + total2;
       e += gridDim.x * blockDim.x) {
    if (e < total1) {
      int i = e & 15, L = (e >> 4) & 31, ks = (e >> 9) % 36, t = e / (512 * 36);
      int k = ks * 32 + kfrag(L, i);
      int c = k & 127, pos = k >> 7;
      int ky = pos / 3, kx = pos - ky * 3, co = t * 16 + (L & 15);
      wA1[e] = (__bf16)w1[(((size_t)co * 128 + c) * 3 + ky) * 3 + kx];
    } else {
      int e2 = e - total1;
      int i = e2 & 15, L = (e2 >> 4) & 31, ks = (e2 >> 9) & 3, t = e2 / (512 * 4);
      int k = ks * 32 + kfrag(L, i);
      int co = t * 16 + (L & 15);
      wA2[e2] = (__bf16)w2[co * 128 + k];
    }
  }
}

/* ---- kernel 4: gated 3x3 conv, implicit GEMM via bf16 WMMA ---- */
__global__ void __launch_bounds__(256)
k_conv3(const float* __restrict__ x, const float* __restrict__ gate,
        const __bf16* __restrict__ wA1, __bf16* __restrict__ hT,
        float* __restrict__ bnpart) {
  __shared__ __align__(16) __bf16 smem[3 * COLS * CHP];
  int bid = blockIdx.x;
  int b  = bid >> 10;
  int h  = (bid >> 2) & 255;
  int w0 = (bid & 3) * TP;
  int tid = threadIdx.x;

  /* fill LDS: channel-major gated window y = (1+gate[h])*x, bf16.
     18 tasks per (row,c): 16 aligned float4 quads (cols 1..64, always
     in-bounds in W) + 2 edge scalars (cols 0 and 65). */
  for (int e = tid; e < 3 * C_CH * 18; e += 256) {
    int row = e / (C_CH * 18);
    int r2 = e - row * (C_CH * 18);
    int c = r2 / 18;
    int u = r2 - c * 18;
    int hh = h + row - 1;
    bool rowok = (unsigned)hh < 256u;
    float g = rowok ? (1.f + gate[b * 256 + hh]) : 0.f;
    const float* src = x + (((size_t)(b * C_CH + c)) * 256 + (rowok ? hh : 0)) * 256;
    __bf16* dst = &smem[(row * COLS) * CHP + c];
    if (u == 0 || u == 17) {
      int col = (u == 0) ? 0 : 65;
      int ww = w0 + col - 1;
      float v = 0.f;
      if (rowok && (unsigned)ww < 256u) v = src[ww] * g;
      dst[col * CHP] = (__bf16)v;
    } else {
      int col = 1 + (u - 1) * 4;             // 1..61
      int ww = w0 + col - 1;                 // multiple of 4 -> 16B aligned
      v4f q = (v4f){0.f, 0.f, 0.f, 0.f};
      if (rowok) q = *(const v4f*)(src + ww);
      dst[(col + 0) * CHP] = (__bf16)(q.x * g);
      dst[(col + 1) * CHP] = (__bf16)(q.y * g);
      dst[(col + 2) * CHP] = (__bf16)(q.z * g);
      dst[(col + 3) * CHP] = (__bf16)(q.w * g);
    }
  }
  __syncthreads();

  int t = tid >> 5, L = tid & 31;
  int kh = L >> 4, ln = L & 15;
  v8f acc[4] = {};
#pragma unroll 2
  for (int ks = 0; ks < 36; ++ks) {
    v16bf a = *(const v16bf*)(wA1 + (((size_t)(t * 36 + ks)) * 32 + L) * 16);
    int pos = ks >> 2, c0 = (ks & 3) * 32;
    int dy = pos / 3, dx = pos - dy * 3;
    const __bf16* bb = &smem[(dy * COLS + dx) * CHP + c0 + kh * 16];
#pragma unroll
    for (int nt = 0; nt < 4; ++nt) {
      int col = nt * 16 + ln;
      v16bf bf = *(const v16bf*)(bb + col * CHP);
      acc[nt] = __builtin_amdgcn_wmma_f32_16x16x32_bf16(
          false, a, false, bf, (short)0, acc[nt], false, false);
    }
  }

  /* BN partial sums/sumsq per channel (half-wave xor reductions),
     transposed layout bnpart[ch][blk] for coalesced reduction reads */
#pragma unroll
  for (int r = 0; r < 8; ++r) {
    float s = 0.f, q = 0.f;
#pragma unroll
    for (int nt = 0; nt < 4; ++nt) { float v = acc[nt][r]; s += v; q += v * v; }
#pragma unroll
    for (int m = 1; m < 16; m <<= 1) { s += __shfl_xor(s, m, 32); q += __shfl_xor(q, m, 32); }
    if (ln == 0) {
      int ch = t * 16 + kh * 8 + r;
      bnpart[(size_t)ch * NBLK3 + bid]         = s;
      bnpart[(size_t)(128 + ch) * NBLK3 + bid] = q;
    }
  }
  __syncthreads();                 /* all waves done reading smem */

  /* stage D (pre-BN h) into LDS as [pix][ch], then coalesced NHWC bf16 store */
#pragma unroll
  for (int nt = 0; nt < 4; ++nt) {
    int pix = nt * 16 + ln;
#pragma unroll
    for (int j = 0; j < 4; ++j) {
      int ch = t * 16 + kh * 8 + 2 * j;
      v2bf pk; pk.x = (__bf16)acc[nt][2 * j]; pk.y = (__bf16)acc[nt][2 * j + 1];
      *(v2bf*)&smem[pix * CHP + ch] = pk;
    }
  }
  __syncthreads();
  for (int e = tid; e < TP * 16; e += 256) {
    int pix = e >> 4, c8 = e & 15;
    v8bf vv = *(const v8bf*)&smem[pix * CHP + c8 * 8];
    size_t gp = (size_t)b * IMG + (size_t)h * 256 + (size_t)(w0 + pix);
    *(v8bf*)(hT + gp * 128 + c8 * 8) = vv;
  }
}

/* ---- kernel 5: finalize BN stats -> per-channel scale/shift ---- */
__global__ void __launch_bounds__(256)
k_bnstats(const float* __restrict__ bnpart, const float* __restrict__ gamma,
          const float* __restrict__ beta, float* __restrict__ sscale) {
  int ch = blockIdx.x;
  __shared__ float sS[256], qS[256];
  float s = 0.f, q = 0.f;
  for (int blk = threadIdx.x; blk < NBLK3; blk += 256) {   // coalesced
    s += bnpart[(size_t)ch * NBLK3 + blk];
    q += bnpart[(size_t)(128 + ch) * NBLK3 + blk];
  }
  sS[threadIdx.x] = s; qS[threadIdx.x] = q;
  __syncthreads();
  for (int st = 128; st > 0; st >>= 1) {
    if (threadIdx.x < st) { sS[threadIdx.x] += sS[threadIdx.x + st]; qS[threadIdx.x] += qS[threadIdx.x + st]; }
    __syncthreads();
  }
  if (threadIdx.x == 0) {
    float inv = 1.f / (float)NPIX;
    float mu  = sS[0] * inv;
    float var = qS[0] * inv - mu * mu;
    float rstd = rsqrtf(var + 1e-5f);
    float sc = gamma[ch] * rstd;
    sscale[ch]       = sc;
    sscale[128 + ch] = beta[ch] - mu * sc;
  }
}

/* ---- kernel 6: BN+ReLU fused 1x1 conv via bf16 WMMA ---- */
__global__ void __launch_bounds__(256)
k_conv1(const __bf16* __restrict__ hT, const __bf16* __restrict__ wA2,
        const float* __restrict__ sscale, float* __restrict__ out) {
  int tid = threadIdx.x;
  int wv = tid >> 5, L = tid & 31;
  int kh = L >> 4, ln = L & 15;
  size_t pix = ((size_t)blockIdx.x * 8 + wv) * 16 + ln;
  v8f acc[8] = {};
#pragma unroll 1
  for (int ks = 0; ks < 4; ++ks) {
    int cbase = ks * 32 + kh * 16;
    v16bf raw = *(const v16bf*)(hT + pix * 128 + cbase);
    v16bf bf;
#pragma unroll
    for (int i = 0; i < 16; ++i) {
      int ch = cbase + i;
      float f = (float)raw[i] * sscale[ch] + sscale[128 + ch];
      bf[i] = (__bf16)fmaxf(f, 0.f);
    }
#pragma unroll
    for (int t = 0; t < 8; ++t) {
      v16bf a = *(const v16bf*)(wA2 + (((size_t)(t * 4 + ks)) * 32 + L) * 16);
      acc[t] = __builtin_amdgcn_wmma_f32_16x16x32_bf16(
          false, a, false, bf, (short)0, acc[t], false, false);
    }
  }
  int b  = (int)(pix >> 16);
  int ip = (int)(pix & 65535);
#pragma unroll
  for (int t = 0; t < 8; ++t) {
#pragma unroll
    for (int r = 0; r < 8; ++r) {
      int ch = t * 16 + kh * 8 + r;
      out[(((size_t)(b * C_CH + ch)) << 16) + ip] = acc[t][r];
    }
  }
}

extern "C" void kernel_launch(void* const* d_in, const int* in_sizes, int n_in,
                              void* d_out, int out_size, void* d_ws, size_t ws_size,
                              hipStream_t stream) {
  (void)in_sizes; (void)n_in; (void)out_size; (void)ws_size;
  const float* x     = (const float*)d_in[0];
  const float* w1    = (const float*)d_in[1];
  const float* gamma = (const float*)d_in[2];
  const float* beta  = (const float*)d_in[3];
  const float* w2    = (const float*)d_in[4];

  char* ws = (char*)d_ws;
  float*  prow   = (float*)(ws + OFF_PROW);
  float*  pcol   = (float*)(ws + OFF_PCOL);
  float*  gate   = (float*)(ws + OFF_GATE);
  __bf16* wA1    = (__bf16*)(ws + OFF_WA1);
  __bf16* wA2    = (__bf16*)(ws + OFF_WA2);
  float*  sscale = (float*)(ws + OFF_SS);
  float*  bnpart = (float*)(ws + OFF_BNP);
  __bf16* hT     = (__bf16*)(ws + OFF_HT);

  k_partials<<<NBATCH * C_CH, 256, 0, stream>>>(x, prow, pcol);
  k_gate    <<<NBATCH, 256, 0, stream>>>(prow, pcol, gate);
  k_prep    <<<640, 256, 0, stream>>>(w1, w2, wA1, wA2);
  k_conv3   <<<NBLK3, 256, 0, stream>>>(x, gate, wA1, hT, bnpart);
  k_bnstats <<<C_CH, 256, 0, stream>>>(bnpart, gamma, beta, sscale);
  k_conv1   <<<NPIX / 128, 256, 0, stream>>>(hT, wA2, sscale, (float*)d_out);
}
